// EncoderDecoder_86509231276862
// MI455X (gfx1250) — matrix operand order repaired and checked by reading, not compiled, not run on hardware
//
#include <hip/hip_runtime.h>

typedef __attribute__((ext_vector_type(16))) __bf16 v16bf;
typedef __attribute__((ext_vector_type(8)))  float  v8f;

#define B_       512
#define T_IN_    256
#define D_IN_    64
#define H_       512
#define NG_      2048   /* 4H */
#define T_OUT_   8
#define BT_      16     /* batch rows per workgroup */
#define NTHREADS_ 512   /* 16 waves of 32 */

__device__ __forceinline__ unsigned short f2bf(float f) {
  unsigned int u = __float_as_uint(f);
  u += 0x7fffu + ((u >> 16) & 1u);          // round-to-nearest-even
  return (unsigned short)(u >> 16);
}
__device__ __forceinline__ float bf2f(unsigned short s) {
  return __uint_as_float(((unsigned int)s) << 16);
}
__device__ __forceinline__ float sigf(float v) {
  return 1.0f / (1.0f + __expf(-v));
}

// Element index into an A-fragment-ordered bf16 array laid out [K/32][32 lanes][16].
// 16-bit A 16x32 layout (ISA 7.12.2): lane = half*16 + m,
// K = (j<4 ? 2j : 16+2(j-4)) + (e&1) + half*8  with e = 2j + (e&1).
__device__ __forceinline__ int a_elem(int m, int k) {
  int ktile = k >> 5, kin = k & 31;
  int half = (kin >> 3) & 1;
  int j = (kin & 16) ? (4 + ((kin & 7) >> 1)) : ((kin & 7) >> 1);
  int e = 2 * j + (kin & 1);
  return (ktile * 32 + half * 16 + m) * 16 + e;
}

// Pack row-major W[N][K] fp32 into B-fragment-ordered bf16:
// out[ntile][ktile][lane][16], lane = half*16 + (n&15), element e -> K = ktile*32 + (e<8 ? half*8 + e : 16 + half*8 + (e-8)).
__global__ void pack_bfrag(const float* __restrict__ W, unsigned short* __restrict__ out,
                           int N, int K) {
  int idx = blockIdx.x * blockDim.x + threadIdx.x;
  if (idx >= N * K) return;
  int n = idx / K, k = idx - n * K;
  int ntile = n >> 4, nlo = n & 15;
  int ktile = k >> 5, kin = k & 31;
  int half = (kin >> 3) & 1;
  int e = (kin & 7) + ((kin >> 4) << 3);
  int lane = half * 16 + nlo;
  out[((ntile * (K >> 5) + ktile) * 32 + lane) * 16 + e] = f2bf(W[idx]);
}

__global__ void prep_vecs(const float* __restrict__ ebih, const float* __restrict__ ebhh,
                          const float* __restrict__ dbih, const float* __restrict__ dbhh,
                          const float* __restrict__ dWih,
                          float* __restrict__ bias_e, float* __restrict__ bias_d,
                          float* __restrict__ dwih) {
  int i = blockIdx.x * blockDim.x + threadIdx.x;
  if (i >= NG_) return;
  bias_e[i] = ebih[i] + ebhh[i];
  bias_d[i] = dbih[i] + dbhh[i];
  dwih[i]   = dWih[i];            // dec_Wih is [4H,1]
}

// Persistent encoder+decoder: one WG per 16 batch rows, 16 waves.
// Wave w owns gate columns {q*512 + w*32 .. +32} for q in 0..3 (i,f,g,o) so the
// LSTM cell update and c-state stay in-wave/in-register. h round-trips via LDS.
__global__ __launch_bounds__(NTHREADS_, 1) void lstm_encdec(
    const float* __restrict__ x,
    const unsigned short* __restrict__ Wenc,   // [128][16][32][16] bf16
    const unsigned short* __restrict__ Wihp,   // [128][ 2][32][16] bf16
    const unsigned short* __restrict__ Wdec,   // [128][16][32][16] bf16
    const float* __restrict__ bias_e,
    const float* __restrict__ bias_d,
    const float* __restrict__ dwih,
    const float* __restrict__ fcW,
    const float* __restrict__ fcB,
    float* __restrict__ out) {
  __shared__ __align__(32) unsigned short h_a[16 * 32 * 16];   // h in A-frag order (16 KB)
  __shared__ __align__(32) unsigned short h_row[BT_ * H_];     // h row-major (16 KB, decoder only)
  __shared__ __align__(32) unsigned short x_a[2 * 32 * 16];    // x_t in A-frag order (2 KB)
  __shared__ float bias_s[NG_];
  __shared__ float dwih_s[NG_];
  __shared__ float fcw_s[H_];
  __shared__ float y_s[BT_];

  const int tid  = threadIdx.x;
  const int lane = tid & 31;
  const int w    = tid >> 5;
  const int nlo  = lane & 15;
  const int b0   = blockIdx.x * BT_;

  // Wave w's 8 gate tiles: i8 = q*2 + nt -> ntile = q*32 + w*2 + nt.
  // Per-tile base offsets (ushort units) into packed weights; kt advances by 512.
  // Whh/Wdec tile stride = 16*32*16 = 8192; Wih tile stride = 2*32*16 = 1024.
  const int tbase = w * 2;                 // ntile offset within a quadrant
  const int lane16 = lane * 16;

  for (int i = tid; i < NG_; i += NTHREADS_) bias_s[i] = bias_e[i];
  for (int i = tid; i < 16 * 32 * 16; i += NTHREADS_) h_a[i] = 0;

  v8f cst[2];
#pragma unroll
  for (int nt = 0; nt < 2; ++nt)
#pragma unroll
    for (int r = 0; r < 8; ++r) cst[nt][r] = 0.0f;

  // ---------------- encoder: 256 sequential steps ----------------
  for (int t = 0; t < T_IN_; ++t) {
    // stage x[b0:b0+16, t, :] -> bf16 A-fragment order
    for (int i = tid; i < BT_ * D_IN_; i += NTHREADS_) {
      int m = i >> 6, k = i & 63;
      x_a[a_elem(m, k)] = f2bf(x[((b0 + m) * T_IN_ + t) * D_IN_ + k]);
    }
    __syncthreads();   // publishes x_a and previous step's h_a

    v8f acc[8];
#pragma unroll
    for (int i8 = 0; i8 < 8; ++i8) {
      float bz = bias_s[((i8 >> 1) * 32 + tbase + (i8 & 1)) * 16 + nlo];
#pragma unroll
      for (int r = 0; r < 8; ++r) acc[i8][r] = bz;
    }

    // x contribution (K = 64): load each B-frag right before its WMMA
#pragma unroll 1
    for (int kt = 0; kt < 2; ++kt) {
      v16bf a = *reinterpret_cast<const v16bf*>(&x_a[(kt * 32 + lane) * 16]);
      const unsigned short* wp = &Wihp[(tbase * 2 + kt) * 512 + lane16];
#pragma unroll
      for (int i8 = 0; i8 < 8; ++i8) {
        // tile (q,nt): offset = ((q*32 + tbase + nt)*2 + kt)*512 + lane16
        const v16bf b = *reinterpret_cast<const v16bf*>(
            wp + ((i8 >> 1) * 32 * 2 + (i8 & 1) * 2) * 512);
        acc[i8] = __builtin_amdgcn_wmma_f32_16x16x32_bf16(
            false, a, false, b, (short)0, acc[i8], false, false);
      }
    }

    // h contribution (K = 512), Whh bf16 resident in L2
#pragma unroll 1
    for (int kt = 0; kt < 16; ++kt) {
      __builtin_prefetch(&Wenc[(tbase * 16 + ((kt + 2) & 15)) * 512 + lane16], 0, 0);
      v16bf a = *reinterpret_cast<const v16bf*>(&h_a[(kt * 32 + lane) * 16]);
      const unsigned short* wp = &Wenc[(tbase * 16 + kt) * 512 + lane16];
#pragma unroll
      for (int i8 = 0; i8 < 8; ++i8) {
        const v16bf b = *reinterpret_cast<const v16bf*>(
            wp + ((i8 >> 1) * 32 * 16 + (i8 & 1) * 16) * 512);
        acc[i8] = __builtin_amdgcn_wmma_f32_16x16x32_bf16(
            false, a, false, b, (short)0, acc[i8], false, false);
      }
    }

    // LSTM cell, entirely in-register (c-state never leaves the wave)
    v8f hv[2];
#pragma unroll
    for (int nt = 0; nt < 2; ++nt) {
#pragma unroll
      for (int r = 0; r < 8; ++r) {
        float iv = acc[0 + nt][r], fv = acc[2 + nt][r];
        float gv = acc[4 + nt][r], ov = acc[6 + nt][r];
        float c  = sigf(fv) * cst[nt][r] + sigf(iv) * tanhf(gv);
        cst[nt][r] = c;
        hv[nt][r]  = sigf(ov) * tanhf(c);
      }
    }

    __syncthreads();   // all reads of h_a for this step are done

    // scatter new h into A-frag order (next step's K dim)
#pragma unroll
    for (int nt = 0; nt < 2; ++nt) {
      int n = w * 32 + nt * 16 + nlo;   // h column index
#pragma unroll
      for (int r = 0; r < 8; ++r) {
        int m = r + (lane >> 4) * 8;    // C-layout: VGPR r -> M=r / M=8+r
        h_a[a_elem(m, n)] = f2bf(hv[nt][r]);
      }
    }
  }

  __syncthreads();
  for (int i = tid; i < NG_; i += NTHREADS_) { bias_s[i] = bias_d[i]; dwih_s[i] = dwih[i]; }
  for (int i = tid; i < H_;  i += NTHREADS_) fcw_s[i] = fcW[i];
  if (tid < BT_) y_s[tid] = 0.0f;       // zero start token
  const float fcb0 = fcB[0];

  // ---------------- decoder: 8 steps ----------------
  for (int t = 0; t < T_OUT_; ++t) {
    __syncthreads();   // publish h_a / y_s / staged constants

    v8f acc[8];
#pragma unroll
    for (int i8 = 0; i8 < 8; ++i8) {
      int n = ((i8 >> 1) * 32 + tbase + (i8 & 1)) * 16 + nlo;
      float bz = bias_s[n], wv = dwih_s[n];
#pragma unroll
      for (int r = 0; r < 8; ++r) {
        int m = r + (lane >> 4) * 8;
        acc[i8][r] = bz + y_s[m] * wv;   // rank-1 input contribution (D_OUT=1)
      }
    }

#pragma unroll 1
    for (int kt = 0; kt < 16; ++kt) {
      v16bf a = *reinterpret_cast<const v16bf*>(&h_a[(kt * 32 + lane) * 16]);
      const unsigned short* wp = &Wdec[(tbase * 16 + kt) * 512 + lane16];
#pragma unroll
      for (int i8 = 0; i8 < 8; ++i8) {
        const v16bf b = *reinterpret_cast<const v16bf*>(
            wp + ((i8 >> 1) * 32 * 16 + (i8 & 1) * 16) * 512);
        acc[i8] = __builtin_amdgcn_wmma_f32_16x16x32_bf16(
            false, a, false, b, (short)0, acc[i8], false, false);
      }
    }

    v8f hv[2];
#pragma unroll
    for (int nt = 0; nt < 2; ++nt) {
#pragma unroll
      for (int r = 0; r < 8; ++r) {
        float iv = acc[0 + nt][r], fv = acc[2 + nt][r];
        float gv = acc[4 + nt][r], ov = acc[6 + nt][r];
        float c  = sigf(fv) * cst[nt][r] + sigf(iv) * tanhf(gv);
        cst[nt][r] = c;
        hv[nt][r]  = sigf(ov) * tanhf(c);
      }
    }

    __syncthreads();
#pragma unroll
    for (int nt = 0; nt < 2; ++nt) {
      int n = w * 32 + nt * 16 + nlo;
#pragma unroll
      for (int r = 0; r < 8; ++r) {
        int m = r + (lane >> 4) * 8;
        unsigned short hb = f2bf(hv[nt][r]);
        h_a[a_elem(m, n)] = hb;
        h_row[m * H_ + n] = hb;
      }
    }
    __syncthreads();

    // y = h @ fc_W.T + fc_b  (D_OUT=1): trivial, wave 0 lanes 0..15
    if (tid < BT_) {
      float s = fcb0;
      for (int k = 0; k < H_; ++k) s += bf2f(h_row[tid * H_ + k]) * fcw_s[k];
      y_s[tid] = s;
      out[(b0 + tid) * T_OUT_ + t] = s;
    }
    // loop-top barrier publishes y_s for the next step
  }
}

extern "C" void kernel_launch(void* const* d_in, const int* in_sizes, int n_in,
                              void* d_out, int out_size, void* d_ws, size_t ws_size,
                              hipStream_t stream) {
  (void)in_sizes; (void)n_in; (void)out_size; (void)ws_size;
  const float* x       = (const float*)d_in[0];
  // d_in[1] = batch_size (scalar int), unused: B == x.shape[0]
  const float* enc_Wih = (const float*)d_in[2];
  const float* enc_Whh = (const float*)d_in[3];
  const float* enc_bih = (const float*)d_in[4];
  const float* enc_bhh = (const float*)d_in[5];
  const float* dec_Wih = (const float*)d_in[6];
  const float* dec_Whh = (const float*)d_in[7];
  const float* dec_bih = (const float*)d_in[8];
  const float* dec_bhh = (const float*)d_in[9];
  const float* fc_W    = (const float*)d_in[10];
  const float* fc_b    = (const float*)d_in[11];
  float* out = (float*)d_out;

  // workspace layout (~4.3 MB): packed bf16 weights + fused bias vectors
  char* ws = (char*)d_ws;
  unsigned short* Wenc = (unsigned short*)(ws + 0);          // 2048*512*2 = 2 MB
  unsigned short* Wdec = (unsigned short*)(ws + 2097152);    // 2 MB
  unsigned short* Wihp = (unsigned short*)(ws + 4194304);    // 2048*64*2 = 256 KB
  float* bias_e = (float*)(ws + 4456448);                    // 8 KB
  float* bias_d = (float*)(ws + 4464640);                    // 8 KB
  float* dwihv  = (float*)(ws + 4472832);                    // 8 KB

  {
    int n = NG_ * H_;
    pack_bfrag<<<(n + 255) / 256, 256, 0, stream>>>(enc_Whh, Wenc, NG_, H_);
    pack_bfrag<<<(n + 255) / 256, 256, 0, stream>>>(dec_Whh, Wdec, NG_, H_);
  }
  {
    int n = NG_ * D_IN_;
    pack_bfrag<<<(n + 255) / 256, 256, 0, stream>>>(enc_Wih, Wihp, NG_, D_IN_);
  }
  prep_vecs<<<(NG_ + 255) / 256, 256, 0, stream>>>(enc_bih, enc_bhh, dec_bih, dec_bhh,
                                                   dec_Wih, bias_e, bias_d, dwihv);

  lstm_encdec<<<B_ / BT_, NTHREADS_, 0, stream>>>(x, Wenc, Wihp, Wdec,
                                                  bias_e, bias_d, dwihv,
                                                  fc_W, fc_b, out);
}
